// EGSC_14620068676119
// MI455X (gfx1250) — matrix operand order repaired and checked by reading, not compiled, not run on hardware
//
#include <hip/hip_runtime.h>

#define DD 128
#define ACT_NONE 0
#define ACT_RELU 1
#define ACT_TANH 2
#define ACT_SIG  3

typedef __attribute__((ext_vector_type(16))) __bf16 v16bf;
typedef __attribute__((ext_vector_type(8)))  float  v8f;

__device__ __forceinline__ float actf(float v, int act) {
  if (act == ACT_RELU) return v > 0.f ? v : 0.f;
  if (act == ACT_TANH) return tanhf(v);
  if (act == ACT_SIG)  return 1.f / (1.f + __expf(-v));
  return v;
}

// ---------------------------------------------------------------------------
// WMMA GEMM (templated on K, DOUT): C[Nrows, DOUT] = act(A[Nrows,K](f32) @ W + b)
// W pre-packed as WT bf16 [DOUT][K] (row n = column n of W, contiguous K).
// A fragments for all K chunks are loaded+converted ONCE per wave and reused
// across all column tiles. Row index is clamped (not exec-masked) so the hot
// loop has no exec manipulation; out-of-range rows are simply never stored.
// Block = 256 threads = 8 waves; each wave owns a 16-row strip; block = 128 rows.
// ---------------------------------------------------------------------------
template <int K, int DOUT>
__global__ void __launch_bounds__(256) gemm_t(
    const float* __restrict__ A, int lda,
    const __bf16* __restrict__ WT,
    const float* __restrict__ bias,
    float* __restrict__ Cmat, int ldc,
    int Nrows, int act)
{
  const int lane    = threadIdx.x & 31;
  const int wave    = threadIdx.x >> 5;
  const int rowTile = blockIdx.x * 128 + wave * 16;
  const int m       = lane & 15;
  int row = rowTile + m;
  if (row >= Nrows) row = Nrows - 1;          // clamp: always-valid loads
  const int khA = (lane < 16) ? 0 : 16;       // A-matrix: contiguous k = 16h + s (matches B)
  const int khB = (lane < 16) ? 0 : 16;       // B-matrix 16-bit layout

  constexpr int NK = K / 32;
  v16bf afrag[NK];
  const float* arow = A + (size_t)row * lda + khA;
#pragma unroll
  for (int kc = 0; kc < NK; ++kc) {
    const float* ap = arow + kc * 32;
    float4 a0 = *(const float4*)(ap);
    float4 a1 = *(const float4*)(ap + 4);
    float4 a2 = *(const float4*)(ap + 8);
    float4 a3 = *(const float4*)(ap + 12);
    v16bf a;
    a[0]  = (__bf16)a0.x; a[1]  = (__bf16)a0.y; a[2]  = (__bf16)a0.z; a[3]  = (__bf16)a0.w;
    a[4]  = (__bf16)a1.x; a[5]  = (__bf16)a1.y; a[6]  = (__bf16)a1.z; a[7]  = (__bf16)a1.w;
    a[8]  = (__bf16)a2.x; a[9]  = (__bf16)a2.y; a[10] = (__bf16)a2.z; a[11] = (__bf16)a2.w;
    a[12] = (__bf16)a3.x; a[13] = (__bf16)a3.y; a[14] = (__bf16)a3.z; a[15] = (__bf16)a3.w;
    afrag[kc] = a;
  }

  const int mbase = rowTile + ((lane < 16) ? 0 : 8);
#pragma unroll
  for (int ct = 0; ct < DOUT / 16; ++ct) {
    const int col = ct * 16 + m;
    const __bf16* wp = WT + (size_t)col * K + khB;
    v8f acc = {};
#pragma unroll
    for (int kc = 0; kc < NK; ++kc) {
      v16bf b = *(const v16bf*)(wp + kc * 32);
      acc = __builtin_amdgcn_wmma_f32_16x16x32_bf16(false, afrag[kc], false, b,
                                                    (short)0, acc, false, false);
    }
    const float bv = bias ? bias[col] : 0.f;
#pragma unroll
    for (int r = 0; r < 8; ++r) {
      int rr = mbase + r;
      if (rr < Nrows) Cmat[(size_t)rr * ldc + col] = actf(acc[r] + bv, act);
    }
  }
}

// ---------------------------------------------------------------------------
// Weight packing: W[R,Cc] fp32 -> WT[Cc,Kpad] bf16 (transposed, K zero-padded)
// ---------------------------------------------------------------------------
__global__ void pack_wt(const float* __restrict__ W, __bf16* __restrict__ WT,
                        int R, int Cc, int Kpad)
{
  int i = blockIdx.x * blockDim.x + threadIdx.x;
  int total = Cc * Kpad;
  if (i >= total) return;
  int c = i / Kpad, r = i - c * Kpad;
  float v = (r < R) ? W[(size_t)r * Cc + c] : 0.f;
  WT[(size_t)c * Kpad + r] = (__bf16)v;
}

// ---------------------------------------------------------------------------
// Elementwise / graph kernels
// ---------------------------------------------------------------------------
__global__ void init_h(const float* __restrict__ x, float* __restrict__ h,
                       const float* __restrict__ eps, int l, int n) {
  int i = blockIdx.x * blockDim.x + threadIdx.x;
  if (i < n) h[i] = (1.f + eps[l]) * x[i];
}

__global__ void edge_agg(const float* __restrict__ x, const int* __restrict__ ei,
                         float* __restrict__ h, int En) {
  int idx = blockIdx.x * blockDim.x + threadIdx.x;
  int e = idx >> 5, lane = idx & 31;
  if (e >= En) return;
  int src = ei[e], dst = ei[En + e];
  float4 v = *(const float4*)(x + (size_t)src * DD + lane * 4);
  float* hp = h + (size_t)dst * DD + lane * 4;
  atomicAdd(hp + 0, v.x); atomicAdd(hp + 1, v.y);
  atomicAdd(hp + 2, v.z); atomicAdd(hp + 3, v.w);
}

__global__ void __launch_bounds__(256) bn_stats(const float* __restrict__ h,
    float* __restrict__ gsum, float* __restrict__ gsumsq, int Nn) {
  __shared__ float s1[256], s2[256];
  int col = threadIdx.x & 127, rp = threadIdx.x >> 7;
  int base = blockIdx.x * 256;
  float s = 0.f, ss = 0.f;
  for (int i = 0; i < 128; ++i) {
    int r = base + rp + 2 * i;
    if (r < Nn) { float v = h[(size_t)r * DD + col]; s += v; ss += v * v; }
  }
  s1[threadIdx.x] = s; s2[threadIdx.x] = ss;
  __syncthreads();
  if (rp == 0) {
    atomicAdd(&gsum[col],   s1[threadIdx.x] + s1[threadIdx.x + 128]);
    atomicAdd(&gsumsq[col], s2[threadIdx.x] + s2[threadIdx.x + 128]);
  }
}

__global__ void bn_final(const float* gsum, const float* gsumsq,
                         const float* __restrict__ gamma, const float* __restrict__ beta,
                         float* scale, float* shift, int Nn) {
  int c = threadIdx.x;
  float mu   = gsum[c] / (float)Nn;
  float var  = gsumsq[c] / (float)Nn - mu * mu;
  float rstd = rsqrtf(var + 1e-5f);
  float sc   = gamma[c] * rstd;
  scale[c] = sc;
  shift[c] = beta[c] - mu * sc;
}

__global__ void bn_apply(const float* __restrict__ h, const float* __restrict__ scale,
                         const float* __restrict__ shift, float* __restrict__ xo, int n) {
  int i = blockIdx.x * blockDim.x + threadIdx.x;
  if (i < n) {
    int c = i & (DD - 1);
    float v = h[i] * scale[c] + shift[c];
    xo[i] = v > 0.f ? v : 0.f;
  }
}

__global__ void zero_f32(float* p, int n) {
  int i = blockIdx.x * blockDim.x + threadIdx.x;
  if (i < n) p[i] = 0.f;
}

__global__ void count_nodes(const int* __restrict__ gidx, float* cnt, int n) {
  int i = blockIdx.x * blockDim.x + threadIdx.x;
  if (i < n) atomicAdd(&cnt[gidx[i]], 1.f);
}

__global__ void scp1_mul(const float* __restrict__ coef, const float* __restrict__ x,
                         float* __restrict__ o, int n) {
  int i = blockIdx.x * blockDim.x + threadIdx.x;
  if (i < n) o[i] = (coef[i] + 1.f) * x[i];
}

__global__ void seg_sum(const float* __restrict__ s, const int* __restrict__ gidx,
                        float* __restrict__ gs, int n) {
  int idx = blockIdx.x * blockDim.x + threadIdx.x;
  int node = idx >> 5, lane = idx & 31;
  if (node >= n) return;
  int g = gidx[node];
  float4 v = *(const float4*)(s + (size_t)node * DD + lane * 4);
  float* p = gs + (size_t)g * DD + lane * 4;
  atomicAdd(p + 0, v.x); atomicAdd(p + 1, v.y);
  atomicAdd(p + 2, v.z); atomicAdd(p + 3, v.w);
}

__global__ void graph_mean(const float* __restrict__ gs, const float* __restrict__ cnt,
                           float* __restrict__ m, int n) {
  int i = blockIdx.x * blockDim.x + threadIdx.x;
  if (i < n) {
    float c = cnt[i >> 7];
    m[i] = (c > 0.f) ? gs[i] / c : 0.f;
  }
}

// fused: node_coef = sigmoid(dot(s[i], gc[g(i)])); pooled[g] += node_coef * s[i]
__global__ void node_pool(const float* __restrict__ s, const float* __restrict__ gc,
                          const int* __restrict__ gidx, float* __restrict__ pooled, int n) {
  int idx = blockIdx.x * blockDim.x + threadIdx.x;
  int node = idx >> 5, lane = idx & 31;
  if (node >= n) return;
  int g = gidx[node];
  float4 sv = *(const float4*)(s  + (size_t)node * DD + lane * 4);
  float4 gv = *(const float4*)(gc + (size_t)g    * DD + lane * 4);
  float d = sv.x * gv.x + sv.y * gv.y + sv.z * gv.z + sv.w * gv.w;
  for (int off = 16; off > 0; off >>= 1) d += __shfl_xor(d, off, 32);
  float nc = 1.f / (1.f + __expf(-d));
  float* p = pooled + (size_t)g * DD + lane * 4;
  atomicAdd(p + 0, nc * sv.x); atomicAdd(p + 1, nc * sv.y);
  atomicAdd(p + 2, nc * sv.z); atomicAdd(p + 3, nc * sv.w);
}

__global__ void comb_build(const float* __restrict__ q, const float* __restrict__ c,
                           float* __restrict__ o, int n) {  // n = B*256
  int i = blockIdx.x * blockDim.x + threadIdx.x;
  if (i < n) {
    int b = i >> 8, f = i & 255;
    o[i] = (f < 128) ? q[b * 128 + f] : c[b * 128 + (f - 128)];
  }
}

__global__ void final_dot(const float* __restrict__ h, const float* __restrict__ w,
                          const float* __restrict__ b, float* __restrict__ out, int Bn) {
  int i = blockIdx.x * blockDim.x + threadIdx.x;
  if (i >= Bn) return;
  float s = b[0];
#pragma unroll 8
  for (int k = 0; k < 64; ++k) s += h[(size_t)i * 64 + k] * w[k];
  out[i] = s;
}

// ---------------------------------------------------------------------------
extern "C" void kernel_launch(void* const* d_in, const int* in_sizes, int n_in,
                              void* d_out, int out_size, void* d_ws, size_t ws_size,
                              hipStream_t stream) {
  (void)n_in; (void)ws_size;
  auto cdiv = [](long long a, long long b) -> int { return (int)((a + b - 1) / b); };

  const int Nn = in_sizes[0] / DD;   // 100000
  const int En = in_sizes[1] / 2;    // 600000
  const int Bn = out_size;           // 128
  int Ln = in_sizes[7];              // 3
  if (Ln > 8) Ln = 8;

  // ---- inputs (setup_inputs dict order) ----
  const float* query_x    = (const float*)d_in[0];
  const int*   q_ei       = (const int*)  d_in[1];
  const int*   q_gidx     = (const int*)  d_in[2];
  const float* corpus_x   = (const float*)d_in[3];
  const int*   c_ei       = (const int*)  d_in[4];
  const int*   c_gidx     = (const int*)  d_in[5];
  const float* gin_eps    = (const float*)d_in[7];
  const float* mlp_W1     = (const float*)d_in[8];
  const float* mlp_b1     = (const float*)d_in[9];
  const float* mlp_W2     = (const float*)d_in[10];
  const float* mlp_b2     = (const float*)d_in[11];
  const float* bn_gamma   = (const float*)d_in[12];
  const float* bn_beta    = (const float*)d_in[13];
  const float* pool_G1    = (const float*)d_in[14];
  const float* pool_g1b   = (const float*)d_in[15];
  const float* pool_G2    = (const float*)d_in[16];
  const float* pool_g2b   = (const float*)d_in[17];
  const float* pool_Wm    = (const float*)d_in[18];
  const float* int_G1     = (const float*)d_in[19];
  const float* int_g1b    = (const float*)d_in[20];
  const float* int_G2     = (const float*)d_in[21];
  const float* int_g2b    = (const float*)d_in[22];
  const float* int_F1     = (const float*)d_in[23];
  const float* int_f1b    = (const float*)d_in[24];
  const float* int_F2     = (const float*)d_in[25];
  const float* int_f2b    = (const float*)d_in[26];
  const float* sc_G1      = (const float*)d_in[27];
  const float* sc_g1b     = (const float*)d_in[28];
  const float* sc_G2      = (const float*)d_in[29];
  const float* sc_g2b     = (const float*)d_in[30];
  const float* sc_F1      = (const float*)d_in[31];
  const float* sc_f1b     = (const float*)d_in[32];
  const float* sc_F2      = (const float*)d_in[33];
  const float* sc_f2b     = (const float*)d_in[34];
  float* outp = (float*)d_out;

  // ---- workspace carve-up ----
  char* base = (char*)d_ws;
  size_t off = 0;
  auto alloc = [&](size_t elems, size_t elsz) -> void* {
    void* p = base + off;
    off += ((elems * elsz + 255) & ~(size_t)255);
    return p;
  };
  float* Xbuf  = (float*)alloc((size_t)Nn * DD, 4);
  float* Abuf  = (float*)alloc((size_t)Nn * DD, 4);
  float* B1buf = (float*)alloc((size_t)Nn * DD, 4);
  float* cnt   = (float*)alloc(Bn, 4);
  float* Gsum  = (float*)alloc((size_t)Bn * DD, 4);
  float* Mbuf  = (float*)alloc((size_t)Bn * DD, 4);
  float* GCbuf = (float*)alloc((size_t)Bn * DD, 4);
  float* pooledQ = (float*)alloc((size_t)Ln * Bn * DD, 4);
  float* pooledC = (float*)alloc((size_t)Ln * Bn * DD, 4);
  float* comb  = (float*)alloc((size_t)Bn * 256, 4);
  float* ic1   = (float*)alloc((size_t)Bn * 64, 4);
  float* icoef = (float*)alloc((size_t)Bn * 256, 4);
  float* isc   = (float*)alloc((size_t)Bn * 256, 4);
  float* it1   = (float*)alloc((size_t)Bn * 256, 4);
  float* feats = (float*)alloc((size_t)Bn * 192, 4);
  float* fg1   = (float*)alloc((size_t)Bn * 64, 4);
  float* fcoef = (float*)alloc((size_t)Bn * 192, 4);
  float* sf    = (float*)alloc((size_t)Bn * 192, 4);
  float* ff1   = (float*)alloc((size_t)Bn * 64, 4);
  float* bnsum   = (float*)alloc(256, 4);            // [0..127] sum, [128..255] sumsq
  float* bnscale = (float*)alloc(DD, 4);
  float* bnshift = (float*)alloc(DD, 4);

  __bf16 *W1T[8], *W2T[8], *G1T[8], *G2T[8], *WmT[8];
  __bf16 *iG1T[8], *iG2T[8], *iF1T[8], *iF2T[8];
  for (int l = 0; l < Ln; ++l) {
    W1T[l]  = (__bf16*)alloc(128 * 128, 2);
    W2T[l]  = (__bf16*)alloc(128 * 128, 2);
    G1T[l]  = (__bf16*)alloc(32 * 128, 2);
    G2T[l]  = (__bf16*)alloc(128 * 32, 2);
    WmT[l]  = (__bf16*)alloc(128 * 128, 2);
    iG1T[l] = (__bf16*)alloc(64 * 256, 2);
    iG2T[l] = (__bf16*)alloc(256 * 64, 2);
    iF1T[l] = (__bf16*)alloc(256 * 256, 2);
    iF2T[l] = (__bf16*)alloc(64 * 256, 2);
  }
  __bf16* scG1T = (__bf16*)alloc(48 * 192, 2);
  __bf16* scG2T = (__bf16*)alloc(192 * 64, 2);   // K=48 padded to 64
  __bf16* scF1T = (__bf16*)alloc(64 * 192, 2);

  auto packW = [&](const float* W, __bf16* WT, int R, int Cc, int Kpad) {
    int tot = Cc * Kpad;
    pack_wt<<<cdiv(tot, 256), 256, 0, stream>>>(W, WT, R, Cc, Kpad);
  };
  for (int l = 0; l < Ln; ++l) {
    packW(mlp_W1  + (size_t)l * 128 * 128, W1T[l],  128, 128, 128);
    packW(mlp_W2  + (size_t)l * 128 * 128, W2T[l],  128, 128, 128);
    packW(pool_G1 + (size_t)l * 128 * 32,  G1T[l],  128, 32,  128);
    packW(pool_G2 + (size_t)l * 32 * 128,  G2T[l],  32,  128, 32);
    packW(pool_Wm + (size_t)l * 128 * 128, WmT[l],  128, 128, 128);
    packW(int_G1  + (size_t)l * 256 * 64,  iG1T[l], 256, 64,  256);
    packW(int_G2  + (size_t)l * 64 * 256,  iG2T[l], 64,  256, 64);
    packW(int_F1  + (size_t)l * 256 * 256, iF1T[l], 256, 256, 256);
    packW(int_F2  + (size_t)l * 256 * 64,  iF2T[l], 256, 64,  256);
  }
  packW(sc_G1, scG1T, 192, 48, 192);
  packW(sc_G2, scG2T, 48, 192, 64);
  packW(sc_F1, scF1T, 192, 64, 192);

  // GEMM dispatch over the (K, Dout) shapes used by this model
  auto gemm = [&](const float* A, int lda, const __bf16* WT, const float* bias,
                  float* Cm, int ldc, int Nr, int K, int Dout, int act) {
    dim3 g(cdiv(Nr, 128)), b(256);
#define GEMM_CASE(KK, DO)                                                       \
    if (K == KK && Dout == DO) {                                                \
      gemm_t<KK, DO><<<g, b, 0, stream>>>(A, lda, WT, bias, Cm, ldc, Nr, act);  \
      return;                                                                   \
    }
    GEMM_CASE(128, 128)
    GEMM_CASE(128, 32)
    GEMM_CASE(32, 128)
    GEMM_CASE(256, 64)
    GEMM_CASE(64, 256)
    GEMM_CASE(256, 256)
    GEMM_CASE(192, 48)
    GEMM_CASE(64, 192)
    GEMM_CASE(192, 64)
#undef GEMM_CASE
  };

  auto process_side = [&](const float* x_in, const int* ei, const int* gidx,
                          float* pooledSide) {
    zero_f32<<<cdiv(Bn, 256), 256, 0, stream>>>(cnt, Bn);
    count_nodes<<<cdiv(Nn, 256), 256, 0, stream>>>(gidx, cnt, Nn);
    const float* xcur = x_in;
    for (int l = 0; l < Ln; ++l) {
      // --- GIN conv ---
      init_h<<<cdiv((long long)Nn * DD, 256), 256, 0, stream>>>(xcur, Abuf, gin_eps, l, Nn * DD);
      edge_agg<<<cdiv((long long)En * 32, 256), 256, 0, stream>>>(xcur, ei, Abuf, En);
      gemm(Abuf, 128, W1T[l], mlp_b1 + (size_t)l * DD, B1buf, 128, Nn, 128, 128, ACT_RELU);
      gemm(B1buf, 128, W2T[l], mlp_b2 + (size_t)l * DD, Abuf, 128, Nn, 128, 128, ACT_NONE);
      // --- BatchNorm (training mode, biased var) + relu ---
      zero_f32<<<1, 256, 0, stream>>>(bnsum, 256);
      bn_stats<<<cdiv(Nn, 256), 256, 0, stream>>>(Abuf, bnsum, bnsum + 128, Nn);
      bn_final<<<1, 128, 0, stream>>>(bnsum, bnsum + 128, bn_gamma + (size_t)l * DD,
                                      bn_beta + (size_t)l * DD, bnscale, bnshift, Nn);
      bn_apply<<<cdiv((long long)Nn * DD, 256), 256, 0, stream>>>(Abuf, bnscale, bnshift,
                                                                  Xbuf, Nn * DD);
      xcur = Xbuf;
      // --- attention pool ---
      gemm(Xbuf, 128, G1T[l], pool_g1b + (size_t)l * 32, B1buf, 128, Nn, 128, 32, ACT_RELU);
      gemm(B1buf, 128, G2T[l], pool_g2b + (size_t)l * DD, Abuf, 128, Nn, 32, 128, ACT_TANH);
      scp1_mul<<<cdiv((long long)Nn * DD, 256), 256, 0, stream>>>(Abuf, Xbuf, B1buf, Nn * DD);
      zero_f32<<<cdiv(Bn * DD, 256), 256, 0, stream>>>(Gsum, Bn * DD);
      seg_sum<<<cdiv((long long)Nn * 32, 256), 256, 0, stream>>>(B1buf, gidx, Gsum, Nn);
      graph_mean<<<cdiv(Bn * DD, 256), 256, 0, stream>>>(Gsum, cnt, Mbuf, Bn * DD);
      gemm(Mbuf, 128, WmT[l], nullptr, GCbuf, 128, Bn, 128, 128, ACT_TANH);
      float* pl = pooledSide + (size_t)l * Bn * DD;
      zero_f32<<<cdiv(Bn * DD, 256), 256, 0, stream>>>(pl, Bn * DD);
      node_pool<<<cdiv((long long)Nn * 32, 256), 256, 0, stream>>>(B1buf, GCbuf, gidx, pl, Nn);
    }
  };

  process_side(query_x,  q_ei, q_gidx, pooledQ);
  process_side(corpus_x, c_ei, c_gidx, pooledC);

  // --- interaction per layer -> feats[B, 192] ---
  for (int l = 0; l < Ln; ++l) {
    comb_build<<<cdiv(Bn * 256, 256), 256, 0, stream>>>(pooledQ + (size_t)l * Bn * DD,
                                                        pooledC + (size_t)l * Bn * DD,
                                                        comb, Bn * 256);
    gemm(comb, 256, iG1T[l], int_g1b + (size_t)l * 64,  ic1,   64,  Bn, 256, 64,  ACT_RELU);
    gemm(ic1,  64,  iG2T[l], int_g2b + (size_t)l * 256, icoef, 256, Bn, 64,  256, ACT_SIG);
    scp1_mul<<<cdiv(Bn * 256, 256), 256, 0, stream>>>(icoef, comb, isc, Bn * 256);
    gemm(isc,  256, iF1T[l], int_f1b + (size_t)l * 256, it1,   256, Bn, 256, 256, ACT_RELU);
    gemm(it1,  256, iF2T[l], int_f2b + (size_t)l * 64,  feats + l * 64, 192, Bn, 256, 64, ACT_RELU);
  }

  // --- final scoring head ---
  zero_f32<<<cdiv(Bn * 64, 256), 256, 0, stream>>>(fg1, Bn * 64);     // pad cols 48..63 = 0
  gemm(feats, 192, scG1T, sc_g1b, fg1,   64,  Bn, 192, 48,  ACT_RELU);
  gemm(fg1,   64,  scG2T, sc_g2b, fcoef, 192, Bn, 64,  192, ACT_SIG);
  scp1_mul<<<cdiv(Bn * 192, 256), 256, 0, stream>>>(fcoef, feats, sf, Bn * 192);
  gemm(sf, 192, scF1T, sc_f1b, ff1, 64, Bn, 192, 64, ACT_RELU);
  final_dot<<<cdiv(Bn, 128), 128, 0, stream>>>(ff1, sc_F2, sc_f2b, outp, Bn);
}
